// RAM_Head_88716844466304
// MI455X (gfx1250) — matrix operand order repaired and checked by reading, not compiled, not run on hardware
//
#include <hip/hip_runtime.h>
#include <hip/hip_bf16.h>
#include <math.h>

typedef __bf16 bf16_t;
typedef __attribute__((ext_vector_type(16))) __bf16 v16bf;
typedef __attribute__((ext_vector_type(8)))  __bf16 v8bf;
typedef __attribute__((ext_vector_type(8)))  float  v8f;

#define HIDC   128
#define CFEAT  256
#define NBATCH 2
#define NPIX   196      // 14*14
#define NPOOL  49       // 7*7
#define W14    14
#define QLEN   12544    // 64*196 per batch
#define KLEN   3136     // 64*49  per batch
#define QTILES ((NBATCH*QLEN)/16)   // 1568
#define PTILES ((NBATCH*KLEN)/16)   // 392

static __device__ __forceinline__ v8f wmma_bf16(v16bf a, v16bf b, v8f c) {
  return __builtin_amdgcn_wmma_f32_16x16x32_bf16(false, a, false, b, (short)0, c,
                                                 false, false);
}

static __device__ __forceinline__ v16bf mk16(v8bf lo, v8bf hi) {
  v16bf r;
#pragma unroll
  for (int e = 0; e < 8; ++e) { r[e] = lo[e]; r[8 + e] = hi[e]; }
  return r;
}

// ---------------- kernel 0: fp32 weights -> bf16 ----------------
__global__ void k_cvt(const float* __restrict__ wq, const float* __restrict__ wk,
                      const float* __restrict__ wv, const float* __restrict__ wre,
                      bf16_t* __restrict__ wqb, bf16_t* __restrict__ wkb,
                      bf16_t* __restrict__ wvb, bf16_t* __restrict__ wreb) {
  int i = blockIdx.x * blockDim.x + threadIdx.x;  // 0..32767
  wqb[i]  = (bf16_t)wq[i];
  wkb[i]  = (bf16_t)wk[i];
  wvb[i]  = (bf16_t)wv[i];
  wreb[i] = (bf16_t)wre[i];
}

// ---------------- kernel 1: Q/K/V projections (maxpool fused for K,V) --------
__global__ void __launch_bounds__(32)
k_qkv(const float* __restrict__ rois,
      const bf16_t* __restrict__ wqb, const bf16_t* __restrict__ wkb,
      const bf16_t* __restrict__ wvb,
      const float* __restrict__ bq, const float* __restrict__ bk,
      const float* __restrict__ bv,
      bf16_t* __restrict__ Qb, bf16_t* __restrict__ Kb, bf16_t* __restrict__ Vtb) {
  const int lane = threadIdx.x;
  const int l16  = lane & 15;
  const int hh   = lane >> 4;
  const int wg   = blockIdx.x;

  if (wg < QTILES) {
    // ---- Q path: 16 full-res pixels per wave ----
    const int g0 = wg * 16;
    v16bf qa[8];
    {
      const int g = g0 + l16;           // A row m = lane%16
      const int n = g / NPIX;
      const int p = g - n * NPIX;
      const float* base = rois + (size_t)n * CFEAT * NPIX + p;
#pragma unroll
      for (int kk = 0; kk < 8; ++kk) {
        v16bf a;
#pragma unroll
        for (int e = 0; e < 16; ++e) {
          int c = kk * 32 + 16 * hh + e;
          a[e] = (bf16_t)base[(size_t)c * NPIX];
        }
        qa[kk] = a;
      }
    }
#pragma unroll
    for (int nt = 0; nt < 8; ++nt) {
      v8f acc = {};
      const int o = nt * 16 + l16;
#pragma unroll
      for (int kk = 0; kk < 8; ++kk) {
        const bf16_t* bp = wqb + (size_t)o * CFEAT + kk * 32 + 16 * hh;
        acc = wmma_bf16(qa[kk], mk16(*(const v8bf*)bp, *(const v8bf*)(bp + 8)), acc);
      }
      const float bias = bq[o];
#pragma unroll
      for (int r = 0; r < 8; ++r) {
        int g = g0 + r + 8 * hh;
        int bb = g / QLEN;
        int qrow = g - bb * QLEN;
        Qb[((size_t)bb * QLEN + qrow) * HIDC + o] = (bf16_t)(acc[r] + bias);
      }
    }
  } else {
    // ---- K/V path: 16 pooled pixels per wave, maxpool fused ----
    const int g0 = (wg - QTILES) * 16;
    v16bf pa[8];
    {
      const int g  = g0 + l16;
      const int n  = g / NPOOL;
      const int pp = g - n * NPOOL;
      const int py = pp / 7, px = pp - 7 * py;
      const float* base = rois + (size_t)n * CFEAT * NPIX + (2 * py * W14 + 2 * px);
#pragma unroll
      for (int kk = 0; kk < 8; ++kk) {
        v16bf a;
#pragma unroll
        for (int e = 0; e < 16; ++e) {
          int c = kk * 32 + 16 * hh + e;
          const float* q = base + (size_t)c * NPIX;
          float m0 = fmaxf(q[0], q[1]);
          float m1 = fmaxf(q[W14], q[W14 + 1]);
          a[e] = (bf16_t)fmaxf(m0, m1);
        }
        pa[kk] = a;
      }
    }
    // K: row-major [key][hid]
#pragma unroll
    for (int nt = 0; nt < 8; ++nt) {
      v8f acc = {};
      const int o = nt * 16 + l16;
#pragma unroll
      for (int kk = 0; kk < 8; ++kk) {
        const bf16_t* bp = wkb + (size_t)o * CFEAT + kk * 32 + 16 * hh;
        acc = wmma_bf16(pa[kk], mk16(*(const v8bf*)bp, *(const v8bf*)(bp + 8)), acc);
      }
      const float bias = bk[o];
#pragma unroll
      for (int r = 0; r < 8; ++r) {
        int g = g0 + r + 8 * hh;
        int bb = g / KLEN;
        int krow = g - bb * KLEN;
        Kb[((size_t)bb * KLEN + krow) * HIDC + o] = (bf16_t)(acc[r] + bias);
      }
    }
    // V: transposed store [hid][key] -> packed 16B store per lane
#pragma unroll
    for (int nt = 0; nt < 8; ++nt) {
      v8f acc = {};
      const int o = nt * 16 + l16;
#pragma unroll
      for (int kk = 0; kk < 8; ++kk) {
        const bf16_t* bp = wvb + (size_t)o * CFEAT + kk * 32 + 16 * hh;
        acc = wmma_bf16(pa[kk], mk16(*(const v8bf*)bp, *(const v8bf*)(bp + 8)), acc);
      }
      const float bias = bv[o];
      int gfirst = g0 + 8 * hh;          // rows r are consecutive keys
      int bb = gfirst / KLEN;
      int kfirst = gfirst - bb * KLEN;
      v8bf pk;
#pragma unroll
      for (int r = 0; r < 8; ++r) pk[r] = (bf16_t)(acc[r] + bias);
      *(v8bf*)(Vtb + ((size_t)bb * HIDC + o) * KLEN + kfirst) = pk;
    }
  }
}

// ---------------- kernel 2: flash attention (no score materialization) ------
__global__ void __launch_bounds__(32)
k_attn(const bf16_t* __restrict__ Qb, const bf16_t* __restrict__ Kb,
       const bf16_t* __restrict__ Vtb, bf16_t* __restrict__ Yb) {
  __shared__ __align__(16) bf16_t P[16][32];
  const int lane = threadIdx.x;
  const int l16 = lane & 15, hh = lane >> 4;
  const int wg = blockIdx.x;               // 0..1567
  const int b = wg / (QLEN / 16);
  const int qrow0 = (wg - b * (QLEN / 16)) * 16;

  // Q A-fragments: 16 rows x 128 (4 K-steps of 32)
  v16bf qa[4];
  {
    const bf16_t* qp = Qb + ((size_t)b * QLEN + qrow0 + l16) * HIDC;
#pragma unroll
    for (int kk = 0; kk < 4; ++kk) {
      const bf16_t* p0 = qp + kk * 32 + 16 * hh;
      const bf16_t* p1 = qp + kk * 32 + 16 * hh + 8;
      qa[kk] = mk16(*(const v8bf*)p0, *(const v8bf*)p1);
    }
  }

  v8f acc[8];
#pragma unroll
  for (int i = 0; i < 8; ++i) acc[i] = (v8f){};
  float mrow[8], lrow[8];
#pragma unroll
  for (int r = 0; r < 8; ++r) { mrow[r] = -INFINITY; lrow[r] = 0.f; }

  const bf16_t* Kbase = Kb + (size_t)b * KLEN * HIDC;
  const bf16_t* Vbase = Vtb + (size_t)b * HIDC * KLEN;

  for (int kc = 0; kc < KLEN; kc += 32) {
    // S = Q x K^T : two 16x16 tiles covering 32 keys
    v8f s0 = {}, s1 = {};
#pragma unroll
    for (int kk = 0; kk < 4; ++kk) {
      const bf16_t* kp = Kbase + (size_t)(kc + l16) * HIDC + kk * 32 + 16 * hh;
      s0 = wmma_bf16(qa[kk], mk16(*(const v8bf*)kp, *(const v8bf*)(kp + 8)), s0);
    }
#pragma unroll
    for (int kk = 0; kk < 4; ++kk) {
      const bf16_t* kp = Kbase + (size_t)(kc + 16 + l16) * HIDC + kk * 32 + 16 * hh;
      s1 = wmma_bf16(qa[kk], mk16(*(const v8bf*)kp, *(const v8bf*)(kp + 8)), s1);
    }
    // online softmax: rows live in 16-lane halves; reduce with xor 1/2/4/8
    float nm[8], sc[8];
#pragma unroll
    for (int r = 0; r < 8; ++r) {
      float cm = fmaxf(s0[r], s1[r]);
      cm = fmaxf(cm, __shfl_xor(cm, 1, 32));
      cm = fmaxf(cm, __shfl_xor(cm, 2, 32));
      cm = fmaxf(cm, __shfl_xor(cm, 4, 32));
      cm = fmaxf(cm, __shfl_xor(cm, 8, 32));
      float newm = fmaxf(mrow[r], cm);
      sc[r] = __expf(mrow[r] - newm);
      nm[r] = newm;
      mrow[r] = newm;
    }
    v8f p0, p1;
#pragma unroll
    for (int r = 0; r < 8; ++r) {
      p0[r] = __expf(s0[r] - nm[r]);
      p1[r] = __expf(s1[r] - nm[r]);
      float rs = p0[r] + p1[r];
      rs += __shfl_xor(rs, 1, 32);
      rs += __shfl_xor(rs, 2, 32);
      rs += __shfl_xor(rs, 4, 32);
      rs += __shfl_xor(rs, 8, 32);
      lrow[r] = lrow[r] * sc[r] + rs;
    }
#pragma unroll
    for (int nt = 0; nt < 8; ++nt)
#pragma unroll
      for (int r = 0; r < 8; ++r) acc[nt][r] *= sc[r];

    // C-layout -> A-layout for P via LDS round trip
#pragma unroll
    for (int r = 0; r < 8; ++r) {
      P[r + 8 * hh][l16]      = (bf16_t)p0[r];
      P[r + 8 * hh][16 + l16] = (bf16_t)p1[r];
    }
    __syncthreads();
    v16bf pfrag;
    {
      const bf16_t* pr = &P[l16][0];
      pfrag = mk16(*(const v8bf*)(pr + 16 * hh), *(const v8bf*)(pr + 16 * hh + 8));
    }
    __syncthreads();

    // acc += P x V  (V pre-transposed -> contiguous B loads)
#pragma unroll
    for (int nt = 0; nt < 8; ++nt) {
      const bf16_t* vp = Vbase + (size_t)(nt * 16 + l16) * KLEN + kc + 16 * hh;
      acc[nt] = wmma_bf16(pfrag, mk16(*(const v8bf*)vp, *(const v8bf*)(vp + 8)),
                          acc[nt]);
    }
  }
  // normalize + store Y bf16 [b][qrow][hid]
#pragma unroll
  for (int nt = 0; nt < 8; ++nt) {
    const int o = nt * 16 + l16;
#pragma unroll
    for (int r = 0; r < 8; ++r) {
      float y = acc[nt][r] / lrow[r];
      Yb[((size_t)b * QLEN + qrow0 + r + 8 * hh) * HIDC + o] = (bf16_t)y;
    }
  }
}

// ---------------- kernel 3: out = rois + bre + wre x Y ----------------------
__global__ void __launch_bounds__(32)
k_proj(const float* __restrict__ rois, const bf16_t* __restrict__ Yb,
       const bf16_t* __restrict__ wreb, const float* __restrict__ bre,
       float* __restrict__ out) {
  const int lane = threadIdx.x;
  const int l16 = lane & 15, hh = lane >> 4;
  const int g0 = blockIdx.x * 16;          // pixel tile over both batches

  // B fragments = Y(hid, pixel): per-lane contiguous hid
  v16bf yb[4];
  {
    int g = g0 + l16;
    int bb = g / QLEN;
    int qrow = g - bb * QLEN;
    const bf16_t* yp = Yb + ((size_t)bb * QLEN + qrow) * HIDC;
#pragma unroll
    for (int kk = 0; kk < 4; ++kk) {
      const bf16_t* p = yp + kk * 32 + 16 * hh;
      yb[kk] = mk16(*(const v8bf*)p, *(const v8bf*)(p + 8));
    }
  }
#pragma unroll 1
  for (int ct = 0; ct < 16; ++ct) {        // 16 tiles of 16 output channels
    v8f acc = {};
    const int co = ct * 16 + l16;          // A row m = lane%16
#pragma unroll
    for (int kk = 0; kk < 4; ++kk) {
      const bf16_t* ap = wreb + (size_t)co * HIDC + kk * 32 + 16 * hh;
      acc = wmma_bf16(mk16(*(const v8bf*)ap, *(const v8bf*)(ap + 8)), yb[kk], acc);
    }
#pragma unroll
    for (int r = 0; r < 8; ++r) {
      int co2 = ct * 16 + r + 8 * hh;
      int g = g0 + l16;                    // C col n = lane%16 -> pixel (coalesced)
      int n = g / NPIX;
      int p = g - n * NPIX;
      size_t idx = ((size_t)n * CFEAT + co2) * NPIX + p;
      out[idx] = rois[idx] + bre[co2] + acc[r];
    }
  }
}

extern "C" void kernel_launch(void* const* d_in, const int* in_sizes, int n_in,
                              void* d_out, int out_size, void* d_ws, size_t ws_size,
                              hipStream_t stream) {
  const float* rois = (const float*)d_in[0];
  // d_in[1] = feature : unused by the reference computation
  const float* wq  = (const float*)d_in[2];
  const float* bq  = (const float*)d_in[3];
  const float* wk  = (const float*)d_in[4];
  const float* bk  = (const float*)d_in[5];
  const float* wv  = (const float*)d_in[6];
  const float* bv  = (const float*)d_in[7];
  const float* wre = (const float*)d_in[8];
  const float* bre = (const float*)d_in[9];
  float* out = (float*)d_out;

  char* ws = (char*)d_ws;
  bf16_t* wqb  = (bf16_t*)ws;  ws += (size_t)HIDC * CFEAT * 2;   // 64 KB
  bf16_t* wkb  = (bf16_t*)ws;  ws += (size_t)HIDC * CFEAT * 2;
  bf16_t* wvb  = (bf16_t*)ws;  ws += (size_t)HIDC * CFEAT * 2;
  bf16_t* wreb = (bf16_t*)ws;  ws += (size_t)CFEAT * HIDC * 2;
  bf16_t* Qb   = (bf16_t*)ws;  ws += (size_t)NBATCH * QLEN * HIDC * 2;  // 6.4 MB
  bf16_t* Kb   = (bf16_t*)ws;  ws += (size_t)NBATCH * KLEN * HIDC * 2;  // 1.6 MB
  bf16_t* Vtb  = (bf16_t*)ws;  ws += (size_t)NBATCH * HIDC * KLEN * 2;  // 1.6 MB
  bf16_t* Yb   = (bf16_t*)ws;  ws += (size_t)NBATCH * QLEN * HIDC * 2;  // 6.4 MB

  k_cvt<<<128, 256, 0, stream>>>(wq, wk, wv, wre, wqb, wkb, wvb, wreb);
  k_qkv<<<QTILES + PTILES, 32, 0, stream>>>(rois, wqb, wkb, wvb, bq, bk, bv,
                                            Qb, Kb, Vtb);
  k_attn<<<QTILES, 32, 0, stream>>>(Qb, Kb, Vtb, Yb);
  k_proj<<<QTILES, 32, 0, stream>>>(rois, Yb, wreb, bre, out);
}
